// Generator_75883482186398
// MI455X (gfx1250) — matrix operand order repaired and checked by reading, not compiled, not run on hardware
//
#include <hip/hip_runtime.h>
#include <hip/hip_bf16.h>
#include <stdint.h>

#define NNODES 16384
#define FTOT   128
#define DTOUCH 64
#define NHID   64
#define NCLASS 32
#define NIN    96
#define NH1    256
#define NH2    64
#define TOPK   70
#define BN_EPS 1e-5f
#define KSEG   4
#define KCH    (NNODES / KSEG)   // 4096 K per segment

typedef __attribute__((ext_vector_type(16))) __bf16        v16bf;
typedef __attribute__((ext_vector_type(2)))  __bf16        v2bf;
typedef __attribute__((ext_vector_type(8)))  float         v8f;
typedef __attribute__((ext_vector_type(8)))  unsigned int  v8u;

// ---- f32 -> packed bf16 pair, round-to-nearest-even ----
__device__ __forceinline__ unsigned pk_bf16(float lo, float hi) {
#if __has_builtin(__builtin_amdgcn_cvt_pk_bf16_f32)
  v2bf r = __builtin_amdgcn_cvt_pk_bf16_f32(lo, hi);
  return __builtin_bit_cast(unsigned, r);
#else
  unsigned a = __float_as_uint(lo);
  unsigned b = __float_as_uint(hi);
  a += 0x7FFFu + ((a >> 16) & 1u);
  b += 0x7FFFu + ((b >> 16) & 1u);
  return (a >> 16) | (b & 0xFFFF0000u);
#endif
}
__device__ __forceinline__ unsigned short bf16_1(float x) {
  unsigned a = __float_as_uint(x);
  a += 0x7FFFu + ((a >> 16) & 1u);
  return (unsigned short)(a >> 16);
}

// ---- K0: one-time adj f32 -> bf16 conversion (row-major, same layout) ----
__global__ __launch_bounds__(256)
void k_cvt(const float* __restrict__ adj, uint2* __restrict__ out) {
  size_t idx = (size_t)blockIdx.x * 256 + threadIdx.x;  // 4 floats per thread
  float4 f = ((const float4*)adj)[idx];
  uint2 o;
  o.x = pk_bf16(f.x, f.y);
  o.y = pk_bf16(f.z, f.w);
  out[idx] = o;
}

// ---- K1: Tt[c][r] = sum_j h[r][j] * w[j][c]   (bf16 transposed output) ----
__global__ __launch_bounds__(256)
void k_hw(const float* __restrict__ h, int ldh,
          const float* __restrict__ w, int dout,
          unsigned short* __restrict__ Tt) {
  int r = blockIdx.x * 256 + threadIdx.x;
  int c = blockIdx.y;
  const float* hr = h + (size_t)r * ldh;
  float acc = 0.f;
#pragma unroll
  for (int j = 0; j < NHID; ++j) acc = fmaf(hr[j], w[j * dout + c], acc);
  Tt[(size_t)c * NNODES + r] = bf16_1(acc);
}

// ---- shared B-fragment loader: lane's output column, K = kstart..+7 and kstart+16..+23 ----
__device__ __forceinline__ v16bf load_bfrag(const unsigned short* __restrict__ Tt,
                                            int col, int kstart) {
  const unsigned short* tc = Tt + (size_t)col * NNODES + kstart;
  uint4 q0 = ((const uint4*)tc)[0];
  uint4 q1 = ((const uint4*)(tc + 16))[0];
  v8u bu;
  bu[0] = q0.x; bu[1] = q0.y; bu[2] = q0.z; bu[3] = q0.w;
  bu[4] = q1.x; bu[5] = q1.y; bu[6] = q1.z; bu[7] = q1.w;
  return __builtin_bit_cast(v16bf, bu);
}

__device__ __forceinline__ void store_ctile(float* __restrict__ S, int ncols,
                                            int rowBase, int half, int colBase,
                                            int m, const v8f& c) {
#pragma unroll
  for (int r = 0; r < 8; ++r)
    S[(size_t)(rowBase + r + 8 * half) * ncols + colBase + m] = c[r];
}

// ---- K2a: Spart[seg] = adj[:, Kseg] @ T[Kseg, :]  (f32 stream + convert) ----
template <int NCT>
__global__ __launch_bounds__(128)
void k_spmm_f32(const float* __restrict__ adj,
                const unsigned short* __restrict__ Tt,
                float* __restrict__ Spart) {
  const int lane = threadIdx.x;
  const int wv   = threadIdx.y;
  const int half = lane >> 4;
  const int m    = lane & 15;
  const int rowBase = (blockIdx.x * 4 + wv) * 16;
  const float* arow = adj + (size_t)(rowBase + m) * NNODES;
  const int kbB = 8 * half;
  const int kbA = 8 * half;
  const int kBeg = blockIdx.y * KCH;
  const int kEnd = kBeg + KCH;
  float* Sp = Spart + (size_t)blockIdx.y * NNODES * (NCT * 16);

  v8f acc[NCT];
#pragma unroll
  for (int j = 0; j < NCT; ++j) acc[j] = v8f{0.f,0.f,0.f,0.f,0.f,0.f,0.f,0.f};

  for (int k0 = kBeg; k0 < kEnd; k0 += 32) {
    if ((k0 & 255) == 0 && k0 + 2048 < kEnd)
      __builtin_prefetch(arow + k0 + 2048, 0, 3);   // global_prefetch_b8
    const float4* p0 = (const float4*)(arow + k0 + kbA);
    const float4* p1 = (const float4*)(arow + k0 + 16 + kbA);
    float4 f0 = p0[0], f1 = p0[1], f2 = p1[0], f3 = p1[1];
    v8u au;
    au[0] = pk_bf16(f0.x, f0.y);  au[1] = pk_bf16(f0.z, f0.w);
    au[2] = pk_bf16(f1.x, f1.y);  au[3] = pk_bf16(f1.z, f1.w);
    au[4] = pk_bf16(f2.x, f2.y);  au[5] = pk_bf16(f2.z, f2.w);
    au[6] = pk_bf16(f3.x, f3.y);  au[7] = pk_bf16(f3.z, f3.w);
    v16bf a = __builtin_bit_cast(v16bf, au);
#pragma unroll
    for (int j = 0; j < NCT; ++j) {
      v16bf b = load_bfrag(Tt, j * 16 + m, k0 + kbB);
      acc[j] = __builtin_amdgcn_wmma_f32_16x16x32_bf16(
          false, a, false, b, (short)0, acc[j], false, false);
    }
  }
#pragma unroll
  for (int j = 0; j < NCT; ++j)
    store_ctile(Sp, NCT * 16, rowBase, half, j * 16, m, acc[j]);
}

// ---- K2b: Spart[seg] = adjBF[:, Kseg] @ T[Kseg, :]  (bf16, half the bytes) ----
template <int NCT>
__global__ __launch_bounds__(128)
void k_spmm_bf(const unsigned short* __restrict__ adjBF,
               const unsigned short* __restrict__ Tt,
               float* __restrict__ Spart) {
  const int lane = threadIdx.x;
  const int wv   = threadIdx.y;
  const int half = lane >> 4;
  const int m    = lane & 15;
  const int rowBase = (blockIdx.x * 4 + wv) * 16;
  const unsigned short* arow = adjBF + (size_t)(rowBase + m) * NNODES;
  const int kbB = 8 * half;
  const int kbA = 8 * half;
  const int kBeg = blockIdx.y * KCH;
  const int kEnd = kBeg + KCH;
  float* Sp = Spart + (size_t)blockIdx.y * NNODES * (NCT * 16);

  v8f acc[NCT];
#pragma unroll
  for (int j = 0; j < NCT; ++j) acc[j] = v8f{0.f,0.f,0.f,0.f,0.f,0.f,0.f,0.f};

  for (int k0 = kBeg; k0 < kEnd; k0 += 32) {
    if ((k0 & 511) == 0 && k0 + 4096 < kEnd)
      __builtin_prefetch(arow + k0 + 4096, 0, 3);   // global_prefetch_b8
    uint4 q0 = *(const uint4*)(arow + k0 + kbA);        // K = kbA..kbA+7
    uint4 q1 = *(const uint4*)(arow + k0 + 16 + kbA);   // K = 16+kbA..+7
    v8u au;
    au[0] = q0.x; au[1] = q0.y; au[2] = q0.z; au[3] = q0.w;
    au[4] = q1.x; au[5] = q1.y; au[6] = q1.z; au[7] = q1.w;
    v16bf a = __builtin_bit_cast(v16bf, au);
#pragma unroll
    for (int j = 0; j < NCT; ++j) {
      v16bf b = load_bfrag(Tt, j * 16 + m, k0 + kbB);
      acc[j] = __builtin_amdgcn_wmma_f32_16x16x32_bf16(
          false, a, false, b, (short)0, acc[j], false, false);
    }
  }
#pragma unroll
  for (int j = 0; j < NCT; ++j)
    store_ctile(Sp, NCT * 16, rowBase, half, j * 16, m, acc[j]);
}

// ---- K2c: S = sum over KSEG partials (fixed order -> deterministic) ----
__global__ __launch_bounds__(256)
void k_reduceS(const float* __restrict__ Spart, float* __restrict__ S,
               size_t total /* NNODES * ncols */) {
  size_t idx = (size_t)blockIdx.x * 256 + threadIdx.x;
  float s = 0.f;
#pragma unroll
  for (int g = 0; g < KSEG; ++g) s += Spart[(size_t)g * total + idx];
  S[idx] = s;
}

// ---- K3a: partial column sums of relu(S + b) (deterministic, no atomics) ----
__global__ __launch_bounds__(256)
void k_stats(const float* __restrict__ S, const float* __restrict__ bias,
             float* __restrict__ partial) {
  __shared__ float ls1[4][NHID];
  __shared__ float ls2[4][NHID];
  const int c = threadIdx.x, ty = threadIdx.y;
  const int r0 = blockIdx.x * 256;
  float s1 = 0.f, s2 = 0.f;
  for (int i = ty; i < 256; i += 4) {
    float u = fmaxf(S[(size_t)(r0 + i) * NHID + c] + bias[c], 0.f);
    s1 += u; s2 += u * u;
  }
  ls1[ty][c] = s1; ls2[ty][c] = s2;
  __syncthreads();
  if (ty == 0) {
    s1 = ls1[0][c] + ls1[1][c] + ls1[2][c] + ls1[3][c];
    s2 = ls2[0][c] + ls2[1][c] + ls2[2][c] + ls2[3][c];
    partial[blockIdx.x * 2 * NHID + c]        = s1;
    partial[blockIdx.x * 2 * NHID + NHID + c] = s2;
  }
}

// ---- K3b: reduce 64 partials -> stats[128] ----
__global__ void k_stats2(const float* __restrict__ partial, float* __restrict__ stats) {
  int t = threadIdx.x;   // 0..127
  float s = 0.f;
  for (int b = 0; b < 64; ++b) s += partial[b * 2 * NHID + t];
  stats[t] = s;
}

// ---- K3c: h_out = batchnorm(relu(S + b)) ----
__global__ __launch_bounds__(256)
void k_bnorm(const float* __restrict__ S, const float* __restrict__ bias,
             const float* __restrict__ stats, float* __restrict__ hout) {
  size_t idx = (size_t)blockIdx.x * 256 + threadIdx.x;
  int c = (int)(idx & (NHID - 1));
  float u   = fmaxf(S[idx] + bias[c], 0.f);
  float mu  = stats[c] * (1.f / NNODES);
  float var = stats[NHID + c] * (1.f / NNODES) - mu * mu;
  hout[idx] = (u - mu) * rsqrtf(var + BN_EPS);
}

// ---- K4: hcat = [relu(S5 + gb5) | x[:,64:128]] ----
__global__ __launch_bounds__(256)
void k_concat(const float* __restrict__ S5, const float* __restrict__ b5,
              const float* __restrict__ x, float* __restrict__ hcat) {
  size_t idx = (size_t)blockIdx.x * 256 + threadIdx.x;
  int r = (int)(idx / NIN), c = (int)(idx - (size_t)r * NIN);
  float v;
  if (c < NCLASS) v = fmaxf(S5[(size_t)r * NCLASS + c] + b5[c], 0.f);
  else            v = x[(size_t)r * FTOT + DTOUCH + (c - NCLASS)];
  hcat[idx] = v;
}

// ---- K5: fused 3-layer MLP, one row per block ----
__global__ __launch_bounds__(256)
void k_mlp(const float* __restrict__ hcat,
           const float* __restrict__ w1, const float* __restrict__ b1,
           const float* __restrict__ w2, const float* __restrict__ b2,
           const float* __restrict__ w3, const float* __restrict__ b3,
           float* __restrict__ scores) {
  __shared__ float row[NIN];
  __shared__ float y1[NH1];
  __shared__ float y2[NH2];
  const int tid = threadIdx.x, r = blockIdx.x;
  if (tid < NIN) row[tid] = hcat[(size_t)r * NIN + tid];
  __syncthreads();
  float acc = b1[tid];
#pragma unroll 4
  for (int k = 0; k < NIN; ++k) acc = fmaf(row[k], w1[k * NH1 + tid], acc);
  y1[tid] = fmaxf(acc, 0.f);
  __syncthreads();
  if (tid < NH2) {
    acc = b2[tid];
#pragma unroll 4
    for (int k = 0; k < NH1; ++k) acc = fmaf(y1[k], w2[k * NH2 + tid], acc);
    y2[tid] = fmaxf(acc, 0.f);
  }
  __syncthreads();
  if (tid == 0) {
    acc = b3[0];
    for (int k = 0; k < NH2; ++k) acc = fmaf(y2[k], w3[k], acc);
    scores[r] = acc;
  }
}

// ---- K6: top-70 by iterative argmax (tie -> lowest index, like lax.top_k) ----
__global__ __launch_bounds__(1024)
void k_topk(const float* __restrict__ scores, float* __restrict__ scratch,
            float* __restrict__ outv, int* __restrict__ outi) {
  __shared__ float sv[1024];
  __shared__ int   si[1024];
  const int tid = threadIdx.x;
  for (int i = tid; i < NNODES; i += 1024) scratch[i] = scores[i];
  __syncthreads();
  for (int t = 0; t < TOPK; ++t) {
    float bv = -__builtin_inff(); int bi = NNODES;
    for (int i = tid; i < NNODES; i += 1024) {
      float v = scratch[i];
      if (v > bv) { bv = v; bi = i; }   // first hit wins -> lowest index
    }
    sv[tid] = bv; si[tid] = bi;
    __syncthreads();
    for (int s = 512; s > 0; s >>= 1) {
      if (tid < s) {
        float ov = sv[tid + s]; int oi = si[tid + s];
        if (ov > sv[tid] || (ov == sv[tid] && oi < si[tid])) { sv[tid] = ov; si[tid] = oi; }
      }
      __syncthreads();
    }
    if (tid == 0) {
      outv[t] = sv[0];
      outi[t] = si[0];
      scratch[si[0]] = -__builtin_inff();
    }
    __syncthreads();
  }
}

extern "C" void kernel_launch(void* const* d_in, const int* in_sizes, int n_in,
                              void* d_out, int out_size, void* d_ws, size_t ws_size,
                              hipStream_t stream) {
  (void)in_sizes; (void)n_in; (void)out_size;
  const float* x   = (const float*)d_in[0];
  const float* adj = (const float*)d_in[1];
  const float* gw[5] = {(const float*)d_in[2], (const float*)d_in[4], (const float*)d_in[6],
                        (const float*)d_in[8], (const float*)d_in[10]};
  const float* gb[5] = {(const float*)d_in[3], (const float*)d_in[5], (const float*)d_in[7],
                        (const float*)d_in[9], (const float*)d_in[11]};
  const float* w1 = (const float*)d_in[12]; const float* b1 = (const float*)d_in[13];
  const float* w2 = (const float*)d_in[14]; const float* b2 = (const float*)d_in[15];
  const float* w3 = (const float*)d_in[16]; const float* b3 = (const float*)d_in[17];

  char* ws = (char*)d_ws;
  size_t off = 0;
  auto alloc = [&](size_t bytes) -> void* {
    void* p = ws + off;
    off += (bytes + 255) & ~(size_t)255;
    return p;
  };
  float*          bufH    = (float*)alloc((size_t)NNODES * NHID * sizeof(float));
  float*          bufS    = (float*)alloc((size_t)NNODES * NHID * sizeof(float));
  float*          Spart   = (float*)alloc((size_t)KSEG * NNODES * NHID * sizeof(float));
  unsigned short* Tt      = (unsigned short*)alloc((size_t)NHID * NNODES * sizeof(unsigned short));
  float*          partial = (float*)alloc((size_t)64 * 2 * NHID * sizeof(float));
  float*          stats   = (float*)alloc((size_t)2 * NHID * sizeof(float));
  float*          hcat    = (float*)alloc((size_t)NNODES * NIN * sizeof(float));
  float*          scores  = (float*)alloc((size_t)NNODES * sizeof(float));
  float*          scratch = (float*)alloc((size_t)NNODES * sizeof(float));

  // If the workspace can hold a bf16 copy of adj (512 MB), convert once per
  // call and halve the dominant HBM stream. ws_size is constant across calls,
  // so this branch is deterministic and graph-capture safe.
  const size_t adjBfBytes = (size_t)NNODES * NNODES * sizeof(unsigned short);
  unsigned short* adjBF = nullptr;
  if (off + adjBfBytes <= ws_size) {
    adjBF = (unsigned short*)alloc(adjBfBytes);
    k_cvt<<<(unsigned)((size_t)NNODES * NNODES / 4 / 256), 256, 0, stream>>>(adj, (uint2*)adjBF);
  }

  const float* hin = x;
  int ldh = FTOT;
  for (int L = 0; L < 5; ++L) {
    const int dout = (L == 4) ? NCLASS : NHID;
    dim3 g1(NNODES / 256, dout);
    k_hw<<<g1, 256, 0, stream>>>(hin, ldh, gw[L], dout, Tt);

    dim3 g2(NNODES / 64, KSEG), b2d(32, 4);
    if (adjBF) {
      if (L == 4) k_spmm_bf<2><<<g2, b2d, 0, stream>>>(adjBF, Tt, Spart);
      else        k_spmm_bf<4><<<g2, b2d, 0, stream>>>(adjBF, Tt, Spart);
    } else {
      if (L == 4) k_spmm_f32<2><<<g2, b2d, 0, stream>>>(adj, Tt, Spart);
      else        k_spmm_f32<4><<<g2, b2d, 0, stream>>>(adj, Tt, Spart);
    }
    const size_t total = (size_t)NNODES * dout;
    k_reduceS<<<(unsigned)(total / 256), 256, 0, stream>>>(Spart, bufS, total);

    if (L < 4) {
      k_stats<<<64, dim3(NHID, 4), 0, stream>>>(bufS, gb[L], partial);
      k_stats2<<<1, 2 * NHID, 0, stream>>>(partial, stats);
      k_bnorm<<<NNODES * NHID / 256, 256, 0, stream>>>(bufS, gb[L], stats, bufH);
      hin = bufH; ldh = NHID;
    }
  }

  k_concat<<<NNODES * NIN / 256, 256, 0, stream>>>(bufS, gb[4], x, hcat);
  k_mlp<<<NNODES, 256, 0, stream>>>(hcat, w1, b1, w2, b2, w3, b3, scores);
  k_topk<<<1, 1024, 0, stream>>>(scores, scratch, (float*)d_out, (int*)d_out + TOPK);
}
